// CoAdaptiveGraphConvolution_73547019977090
// MI455X (gfx1250) — matrix operand order repaired and checked by reading, not compiled, not run on hardware
//
#include <hip/hip_runtime.h>
#include <hip/hip_bf16.h>

// ---------------------------------------------------------------------------
// CoAdaptiveGraphConvolution for MI455X (gfx1250, wave32, WMMA bf16)
//
// Exact algebraic simplification: softmax over w row-sums to 1, so the
// attention branch collapses to per-(i,v) scalars rA[i][v] = 1 + sum_w Afull.
// hidden = sum_i rA[i][v] * (g_w[i] @ x) + sum_i g_b[i];  out = relu(BN(hidden)+x)
// Two passes over x (stats pass + finalize pass), bf16 WMMA with f32 accum.
// ---------------------------------------------------------------------------

typedef __attribute__((ext_vector_type(16))) __bf16 v16bf;
typedef __attribute__((ext_vector_type(8)))  float  v8f;
typedef __attribute__((ext_vector_type(4)))  unsigned int u32x4;

#define N_    128
#define C_    64
#define T_    256
#define V_    25
#define TV    6400           // T_*V_
#define NTV   819200         // N_*T_*V_
#define BN_EPS 1e-5f

// workspace layout (float offsets unless noted)
#define WS_RA     0          // 75 floats: rA[i][v]
#define WS_BT     80         // 64 floats: summed g_b
#define WS_SUM    144        // 64 floats: per-channel sum(hidden)
#define WS_SQ     208        // 64 floats: per-channel sum(hidden^2)
#define WS_SCALE  272        // 64 floats
#define WS_SHIFT  336        // 64 floats
#define WS_GWPACK_BYTES 1600 // __bf16[12288] : WMMA-A-layout packed g_w

// ---------------------------------------------------------------------------
// Kernel 1: tiny prep — rA, bias, zero stats, pack g_w into the WMMA 16-bit
// A-matrix lane layout used here: lane L holds M=L%16, K-half = L/16;
// frag element e: Kfrag = 16*kh + e (contiguous K order, matches B loads).
// Packed so each lane's 16 bf16 are contiguous -> 2x global_load_b128.
// ---------------------------------------------------------------------------
__global__ void prep_kernel(const float* __restrict__ A, const float* __restrict__ ga,
                            const float* __restrict__ g_w, const float* __restrict__ g_b,
                            float* ws, __bf16* gwPack) {
  const int t = threadIdx.x;
  if (t < 75) {
    const int i = t / 25, v = t % 25;
    const float* a = A  + (i * 25 + v) * 25;
    const float* g = ga + (i * 25 + v) * 25;
    float s = 1.0f;                       // softmax row-sum == 1
    for (int w = 0; w < 25; ++w) s += a[w] + g[w];
    ws[WS_RA + t] = s;
  }
  if (t < 64) {
    ws[WS_BT  + t] = g_b[t] + g_b[64 + t] + g_b[128 + t];
    ws[WS_SUM + t] = 0.0f;
    ws[WS_SQ  + t] = 0.0f;
  }
  for (int idx = t; idx < 3 * 2 * 4 * 32 * 16; idx += 256) {
    const int e    = idx & 15;
    const int lane = (idx >> 4) & 31;
    const int mt   = (idx >> 9) & 3;
    const int ks   = (idx >> 11) & 1;
    const int i    = idx >> 12;
    const int m    = mt * 16 + (lane & 15);
    const int kh   = lane >> 4;
    const int j    = e >> 1, p = e & 1;
    const int kf   = 16 * kh + 2 * j + p;
    const int c    = ks * 32 + kf;
    gwPack[idx] = (__bf16)g_w[(i * 64 + m) * 64 + c];
  }
}

// ---------------------------------------------------------------------------
// Kernels 2 & 4: GEMM over columns (t*25+v) per batch n; 64x32 tile / iter.
// 8 waves: wave = mt (M-tile 0..3) x colSub (0..1). 3 branch accumulators.
// ---------------------------------------------------------------------------
template <bool FINAL>
__global__ __launch_bounds__(256)
void gcn_gemm_kernel(const float* __restrict__ x, const float* ws,
                     const __bf16* __restrict__ gwPack,
                     float* sums, float* sumsq, float* __restrict__ out) {
  __shared__ __align__(16) __bf16 xbf[32][72];  // [colLocal][c], padded rows (144B)
  __shared__ float sRA[80];
  __shared__ float sB[64];
  __shared__ float sScale[64];
  __shared__ float sShift[64];
  __shared__ float redS[64];
  __shared__ float redQ[64];

  const int tid    = threadIdx.x;
  const int lane   = tid & 31;
  const int wave   = tid >> 5;
  const int mt     = wave & 3;
  const int colSub = wave >> 2;

  const int n        = blockIdx.x / 50;
  const int tg       = blockIdx.x % 50;
  const int colBase0 = tg * 128;
  const int xn       = n * (C_ * TV);

  if (tid < 75) sRA[tid] = ws[WS_RA + tid];
  if (tid < 64) {
    sB[tid] = ws[WS_BT + tid];
    if (FINAL) { sScale[tid] = ws[WS_SCALE + tid]; sShift[tid] = ws[WS_SHIFT + tid]; }
  }

  // Hoisted A-fragments: 3 branches x 2 K-steps, 32B contiguous per lane.
  v16bf afrag[3][2];
#pragma unroll
  for (int i = 0; i < 3; ++i)
#pragma unroll
    for (int ks = 0; ks < 2; ++ks) {
      const u32x4* p = (const u32x4*)(gwPack + ((((i * 2 + ks) * 4 + mt) * 32) + lane) * 16);
      u32x4 lo = p[0], hi = p[1];
      v16bf a;
      ((u32x4*)&a)[0] = lo; ((u32x4*)&a)[1] = hi;
      afrag[i][ks] = a;
    }

  const int nloc = colSub * 16 + (lane & 15);   // B/C/D column within 32-col tile
  const int kh   = lane >> 4;

  for (int it = 0; it < 4; ++it) {
    const int colBase = colBase0 + it * 32;
    if (!FINAL && tid < 64) { redS[tid] = 0.0f; redQ[tid] = 0.0f; }

    // Load 64(c) x 32(col) fp32 tile; convert channel pairs with packed bf16
    // cvt and store one b32 per pair into the transposed LDS tile [col][c].
#pragma unroll
    for (int r = 0; r < 4; ++r) {
      const int pe  = r * 256 + tid;     // pair index 0..1023
      const int col = pe & 31;
      const int c   = (pe >> 5) * 2;     // even channel of the pair
      const float v0 = x[xn + c * TV + colBase + col];
      const float v1 = x[xn + (c + 1) * TV + colBase + col];
      union { unsigned int u; __bf16 h[2]; } pk;
      pk.h[0] = (__bf16)v0;
      pk.h[1] = (__bf16)v1;
      *(unsigned int*)&xbf[col][c] = pk.u;   // byte offset col*144 + 2c: 4B aligned
    }
    if (it < 3)  // pull next tile toward GL2 (global_prefetch_b8)
      __builtin_prefetch(&x[xn + (tid >> 5) * TV + colBase + 32 + (tid & 31)], 0, 1);
    __syncthreads();

    // 16x16x32 bf16 WMMA, K = 64 in two steps, f32 accumulation.
    v8f acc0 = {}, acc1 = {}, acc2 = {};
#pragma unroll
    for (int ks = 0; ks < 2; ++ks) {
      const u32x4* bp = (const u32x4*)(&xbf[nloc][ks * 32 + kh * 16]);
      u32x4 blo = bp[0], bhi = bp[1];
      v16bf b;
      ((u32x4*)&b)[0] = blo; ((u32x4*)&b)[1] = bhi;
      acc0 = __builtin_amdgcn_wmma_f32_16x16x32_bf16(false, afrag[0][ks], false, b, (short)0, acc0, false, false);
      acc1 = __builtin_amdgcn_wmma_f32_16x16x32_bf16(false, afrag[1][ks], false, b, (short)0, acc1, false, false);
      acc2 = __builtin_amdgcn_wmma_f32_16x16x32_bf16(false, afrag[2][ks], false, b, (short)0, acc2, false, false);
    }

    // Epilogue: combine branches with per-v scalars (col is per-lane constant
    // in the C/D layout), then either accumulate BN stats or finalize.
    const int   colg  = colBase + nloc;
    const int   v25   = colg % 25;
    const float r0 = sRA[v25], r1 = sRA[25 + v25], r2 = sRA[50 + v25];
    const int   obase = mt * 16 + kh * 8;
#pragma unroll
    for (int r = 0; r < 8; ++r) {
      const int o = obase + r;                       // C/D: VGPR r -> M = r + 8*(lane/16)
      float h = r0 * acc0[r] + r1 * acc1[r] + r2 * acc2[r] + sB[o];
      if (FINAL) {
        const float xv = x[xn + o * TV + colg];      // identity residual (C_IN==C_OUT)
        const float y  = h * sScale[o] + sShift[o] + xv;
        out[xn + o * TV + colg] = fmaxf(y, 0.0f);
      } else {
        float p = h, q = h * h;
#pragma unroll
        for (int m = 1; m < 16; m <<= 1) {           // reduce over 16 cols sharing o
          p += __shfl_xor(p, m, 32);
          q += __shfl_xor(q, m, 32);
        }
        if ((lane & 15) == 0) {
          atomicAdd(&redS[o], p);
          atomicAdd(&redQ[o], q);
        }
      }
    }
    __syncthreads();
    if (!FINAL) {
      if (tid < 64) {
        atomicAdd(&sums[tid],  redS[tid]);
        atomicAdd(&sumsq[tid], redQ[tid]);
      }
      __syncthreads();
    }
  }
}

// ---------------------------------------------------------------------------
// Kernel 3: fold stats into BN affine scale/shift.
// ---------------------------------------------------------------------------
__global__ void bn_stats_kernel(const float* __restrict__ gamma,
                                const float* __restrict__ beta, float* ws) {
  const int o = threadIdx.x;
  if (o < 64) {
    const float inv  = 1.0f / (float)NTV;
    const float mean = ws[WS_SUM + o] * inv;
    float var        = ws[WS_SQ + o] * inv - mean * mean;
    var = fmaxf(var, 0.0f);
    const float sc = gamma[o] * rsqrtf(var + BN_EPS);
    ws[WS_SCALE + o] = sc;
    ws[WS_SHIFT + o] = beta[o] - mean * sc;
  }
}

extern "C" void kernel_launch(void* const* d_in, const int* in_sizes, int n_in,
                              void* d_out, int out_size, void* d_ws, size_t ws_size,
                              hipStream_t stream) {
  const float* x     = (const float*)d_in[0];
  const float* A     = (const float*)d_in[1];
  const float* ga    = (const float*)d_in[2];
  // d_in[3..6] (a_w,a_b,b_w,b_b) provably do not affect the output.
  const float* g_w   = (const float*)d_in[7];
  const float* g_b   = (const float*)d_in[8];
  const float* gamma = (const float*)d_in[9];
  const float* beta  = (const float*)d_in[10];

  float*  out    = (float*)d_out;
  float*  ws     = (float*)d_ws;
  __bf16* gwPack = (__bf16*)((char*)d_ws + WS_GWPACK_BYTES);

  prep_kernel<<<1, 256, 0, stream>>>(A, ga, g_w, g_b, ws, gwPack);
  gcn_gemm_kernel<false><<<N_ * 50, 256, 0, stream>>>(x, ws, gwPack,
                                                      ws + WS_SUM, ws + WS_SQ, out);
  bn_stats_kernel<<<1, 64, 0, stream>>>(gamma, beta, ws);
  gcn_gemm_kernel<true><<<N_ * 50, 256, 0, stream>>>(x, ws, gwPack,
                                                     ws + WS_SUM, ws + WS_SQ, out);
}
